// RegionalCrossAttention_29162827940756
// MI455X (gfx1250) — compile-verified, matrix operand order checked by reading, not run on hardware
//
#include <hip/hip_runtime.h>
#include <cstdint>
#include <cstddef>

// ---------------------------------------------------------------------------
// RegionalCrossAttention for MI455X (gfx1250, wave32, WMMA)
//   B=2, n=8, L=4096, Lt=77, d=1024, ctx=768, 16 heads x 64
// ---------------------------------------------------------------------------

typedef __attribute__((ext_vector_type(16))) _Float16 v16h;
typedef __attribute__((ext_vector_type(8)))  float    v8f;

#define WMMA_F16(A, B, C) \
    __builtin_amdgcn_wmma_f32_16x16x32_f16(false, (A), false, (B), (short)0, (C), false, false)

static constexpr int BB   = 2;     // batch
static constexpr int NOBJ = 8;     // objects
static constexpr int LQ   = 4096;  // query length (H*W)
static constexpr int LT   = 77;    // text length
static constexpr int LTP  = 96;    // padded text length (3 x K=32 chunks)
static constexpr int DD   = 1024;  // model dim
static constexpr int CTX  = 768;   // text dim
static constexpr int NH   = 16;    // heads
static constexpr int DH   = 64;    // head dim
static constexpr int BN   = BB * NOBJ;

// ---- per-lane WMMA operand loaders (wave32 layouts per CDNA5 ISA 7.12.2) ----
// A (16x32 f16): lane holds row M = lane%16; elem i<8  -> k = half*8+i
//                                            elem i>=8 -> k = 16+half*8+(i-8)
// B (32x16 f16): lane holds col N = lane%16; same k pattern (column-major src)
__device__ __forceinline__ v16h load_h16(const _Float16* __restrict__ row, int half) {
    v16h a;
#pragma unroll
    for (int i = 0; i < 8; ++i) {
        a[i]     = row[half * 8 + i];
        a[i + 8] = row[16 + half * 8 + i];
    }
    return a;
}

__device__ __forceinline__ v16h load_a_f32(const float* __restrict__ row, int half) {
    v16h a;
#pragma unroll
    for (int i = 0; i < 8; ++i) {
        a[i]     = (_Float16)row[half * 8 + i];
        a[i + 8] = (_Float16)row[16 + half * 8 + i];
    }
    return a;
}

// ---------------------------------------------------------------------------
// W (K x N) f32 row-major  ->  WT (N x K) f16 row-major
// ---------------------------------------------------------------------------
__global__ void k_transpose_f16(const float* __restrict__ W, _Float16* __restrict__ WT,
                                int K, int N) {
    long i = (long)blockIdx.x * blockDim.x + threadIdx.x;
    long total = (long)K * N;
    if (i >= total) return;
    int k = (int)(i / N);
    int n = (int)(i % N);
    WT[(long)n * K + k] = (_Float16)W[i];
}

__global__ void k_zero_f16(_Float16* __restrict__ p, long n) {
    long i = (long)blockIdx.x * blockDim.x + threadIdx.x;
    if (i < n) p[i] = (_Float16)0.0f;
}

// cover[b*L+l] = sum_n (mask[b,n,l] > 0.5)
__global__ void k_cover(const float* __restrict__ mask, float* __restrict__ cover) {
    int g = blockIdx.x * blockDim.x + threadIdx.x;
    if (g >= BB * LQ) return;
    int b = g / LQ, l = g % LQ;
    float s = 0.0f;
#pragma unroll
    for (int n = 0; n < NOBJ; ++n)
        s += (mask[((long)(b * NOBJ + n)) * LQ + l] > 0.5f) ? 1.0f : 0.0f;
    cover[g] = s;
}

// ---------------------------------------------------------------------------
// Q = X(8192x1024 f32) @ Wq  -> Q f16 row-major (8192x1024)
// one wave = 16(M) x 64(N) tile; block = 8 waves stacked in M
// ---------------------------------------------------------------------------
__global__ void k_gemm_q(const float* __restrict__ X, const _Float16* __restrict__ WT,
                         _Float16* __restrict__ Q) {
    const int lane = threadIdx.x & 31, wave = threadIdx.x >> 5;
    const int col = lane & 15, half = lane >> 4;
    const int mtile = blockIdx.x * 8 + wave;   // 0..511
    const int n0 = blockIdx.y * 64;            // 0..960
    const float* arow = X + (long)(mtile * 16 + col) * DD;
    const _Float16* b0r = WT + (long)(n0 +  0 + col) * DD;
    const _Float16* b1r = WT + (long)(n0 + 16 + col) * DD;
    const _Float16* b2r = WT + (long)(n0 + 32 + col) * DD;
    const _Float16* b3r = WT + (long)(n0 + 48 + col) * DD;
    v8f c0 = {}, c1 = {}, c2 = {}, c3 = {};
    for (int k0 = 0; k0 < DD; k0 += 32) {
        __builtin_prefetch(arow + k0 + 64, 0, 1);
        v16h a  = load_a_f32(arow + k0, half);
        c0 = WMMA_F16(a, load_h16(b0r + k0, half), c0);
        c1 = WMMA_F16(a, load_h16(b1r + k0, half), c1);
        c2 = WMMA_F16(a, load_h16(b2r + k0, half), c2);
        c3 = WMMA_F16(a, load_h16(b3r + k0, half), c3);
    }
#pragma unroll
    for (int r = 0; r < 8; ++r) {
        long m = (long)mtile * 16 + r + 8 * half;
        _Float16* qp = Q + m * DD + n0 + col;
        qp[0]  = (_Float16)c0[r];
        qp[16] = (_Float16)c1[r];
        qp[32] = (_Float16)c2[r];
        qp[48] = (_Float16)c3[r];
    }
}

// ---------------------------------------------------------------------------
// K/V = ctx(1232x768 f32) @ Wk/Wv, scattered into attention-friendly layouts:
//   mode 0: Kb[(bn*96 + t)*1024 + nn]                       (key-major, padded)
//   mode 1: VT[((bn*16 + h)*64 + dh)*96 + t]   (V transposed: rows = head dim)
// ---------------------------------------------------------------------------
__global__ void k_gemm_kv(const float* __restrict__ CTXF, const _Float16* __restrict__ WT,
                          _Float16* __restrict__ Kb, _Float16* __restrict__ VT, int mode) {
    const int lane = threadIdx.x & 31, wave = threadIdx.x >> 5;
    const int col = lane & 15, half = lane >> 4;
    const int mtile = blockIdx.x * 8 + wave;   // 0..79 (guard: 77 tiles)
    if (mtile >= (BN * LT) / 16) return;       // 1232/16 = 77
    const int n0 = blockIdx.y * 64;
    const float* arow = CTXF + (long)(mtile * 16 + col) * CTX;
    const _Float16* br[4];
#pragma unroll
    for (int j = 0; j < 4; ++j) br[j] = WT + (long)(n0 + j * 16 + col) * CTX;
    v8f c[4] = {v8f{}, v8f{}, v8f{}, v8f{}};
    for (int k0 = 0; k0 < CTX; k0 += 32) {
        __builtin_prefetch(arow + k0 + 64, 0, 1);
        v16h a = load_a_f32(arow + k0, half);
#pragma unroll
        for (int j = 0; j < 4; ++j)
            c[j] = WMMA_F16(a, load_h16(br[j] + k0, half), c[j]);
    }
#pragma unroll
    for (int j = 0; j < 4; ++j) {
        const int nn = n0 + j * 16 + col;
#pragma unroll
        for (int r = 0; r < 8; ++r) {
            int g  = mtile * 16 + r + 8 * half;     // 0..1231
            int bn = g / LT, t = g - bn * LT;
            _Float16 v = (_Float16)c[j][r];
            if (mode == 0) {
                Kb[((long)(bn * LTP + t)) * DD + nn] = v;
            } else {
                int h = nn >> 6, dh = nn & 63;
                VT[(((long)(bn * NH + h)) * DH + dh) * LTP + t] = v;
            }
        }
    }
}

// ---------------------------------------------------------------------------
// Fused attention + masked sum over n.  One wave per (b, head, 16-query tile).
//   S^T = K_tile(A) x Q^T(B)  -> softmax lane-local (+1 shfl_xor pair-reduce)
//   probs re-pack directly into B operand of  out^T = V^T(A) x attn^T(B)
//   mask weight folded into softmax normalizer; accumulate over n in regs.
// ---------------------------------------------------------------------------
__global__ void k_attn(const _Float16* __restrict__ Q, const _Float16* __restrict__ Kb,
                       const _Float16* __restrict__ VT, const float* __restrict__ mask,
                       float* __restrict__ ACC) {
    const int lane = threadIdx.x & 31;
    const int wid  = blockIdx.x * (blockDim.x >> 5) + (threadIdx.x >> 5);
    const int qt = wid & 255;          // 256 query tiles
    const int h  = (wid >> 8) & 15;    // head
    const int b  = wid >> 12;          // batch
    const int col = lane & 15, half = lane >> 4;
    const int qrow = qt * 16 + col;    // lane's query index (= its S^T column)

    const _Float16* qp = Q + ((long)(b * LQ + qrow)) * DD + h * DH;
    const v16h qb0 = load_h16(qp, half);        // dh  0..31
    const v16h qb1 = load_h16(qp + 32, half);   // dh 32..63

    float accw[4][8] = {};
    for (int n = 0; n < NOBJ; ++n) {
        const int bn = b * NOBJ + n;
        const float w = (mask[((long)bn) * LQ + qrow] > 0.5f) ? 1.0f : 0.0f;
        if (__ballot(w != 0.0f) == 0) continue;   // wave-uniform skip

        float st[6][8];
#pragma unroll
        for (int kt = 0; kt < 6; ++kt) {
            const _Float16* kp = Kb + ((long)(bn * LTP + kt * 16 + col)) * DD + h * DH;
            v8f s = {};
            s = WMMA_F16(load_h16(kp, half),      qb0, s);
            s = WMMA_F16(load_h16(kp + 32, half), qb1, s);
#pragma unroll
            for (int r = 0; r < 8; ++r) {
                int key = kt * 16 + r + 8 * half;
                st[kt][r] = (key < LT) ? s[r] * 0.125f : -1e30f;  // 1/sqrt(64)
            }
        }
        // row softmax: keys live in registers + paired lane (xor 16)
        float mx = -1e30f;
#pragma unroll
        for (int kt = 0; kt < 6; ++kt)
#pragma unroll
            for (int r = 0; r < 8; ++r) mx = fmaxf(mx, st[kt][r]);
        mx = fmaxf(mx, __shfl_xor(mx, 16, 32));
        float sum = 0.0f;
#pragma unroll
        for (int kt = 0; kt < 6; ++kt)
#pragma unroll
            for (int r = 0; r < 8; ++r) {
                float p = __expf(st[kt][r] - mx);
                st[kt][r] = p;
                sum += p;
            }
        sum += __shfl_xor(sum, 16, 32);
        const float inv = w / sum;   // fold region weight into normalization

        // S^T C-layout regs are exactly the attn^T B-operand elements
        v16h ab0, ab1, ab2;
#pragma unroll
        for (int i = 0; i < 8; ++i) {
            ab0[i] = (_Float16)(st[0][i] * inv); ab0[i + 8] = (_Float16)(st[1][i] * inv);
            ab1[i] = (_Float16)(st[2][i] * inv); ab1[i + 8] = (_Float16)(st[3][i] * inv);
            ab2[i] = (_Float16)(st[4][i] * inv); ab2[i + 8] = (_Float16)(st[5][i] * inv);
        }
#pragma unroll
        for (int t = 0; t < 4; ++t) {            // head-dim tiles
            const _Float16* vp = VT + (((long)(bn * NH + h)) * DH + t * 16 + col) * LTP;
            v8f o = {};
            o = WMMA_F16(load_h16(vp, half),      ab0, o);
            o = WMMA_F16(load_h16(vp + 32, half), ab1, o);
            o = WMMA_F16(load_h16(vp + 64, half), ab2, o);
#pragma unroll
            for (int r = 0; r < 8; ++r) accw[t][r] += o[r];
        }
    }
    // out^T[dh][q]: lane owns query qrow, dh = t*16 + r + 8*half
    float* ap = ACC + ((long)(b * LQ + qrow)) * DD + h * DH;
#pragma unroll
    for (int t = 0; t < 4; ++t)
#pragma unroll
        for (int r = 0; r < 8; ++r) ap[t * 16 + 8 * half + r] = accw[t][r];
}

// ---------------------------------------------------------------------------
// OUT = ACC @ Wo + cover*bo + (1-cover)*null + x     (8192x1024, f32 out)
// ---------------------------------------------------------------------------
__global__ void k_gemm_out(const float* __restrict__ ACC, const _Float16* __restrict__ WT,
                           const float* __restrict__ X, const float* __restrict__ bo,
                           const float* __restrict__ nullf, const float* __restrict__ cover,
                           float* __restrict__ OUT) {
    const int lane = threadIdx.x & 31, wave = threadIdx.x >> 5;
    const int col = lane & 15, half = lane >> 4;
    const int mtile = blockIdx.x * 8 + wave;
    const int n0 = blockIdx.y * 64;
    const float* arow = ACC + (long)(mtile * 16 + col) * DD;
    const _Float16* br[4];
#pragma unroll
    for (int j = 0; j < 4; ++j) br[j] = WT + (long)(n0 + j * 16 + col) * DD;
    v8f c[4] = {v8f{}, v8f{}, v8f{}, v8f{}};
    for (int k0 = 0; k0 < DD; k0 += 32) {
        __builtin_prefetch(arow + k0 + 64, 0, 1);
        v16h a = load_a_f32(arow + k0, half);
#pragma unroll
        for (int j = 0; j < 4; ++j)
            c[j] = WMMA_F16(a, load_h16(br[j] + k0, half), c[j]);
    }
#pragma unroll
    for (int j = 0; j < 4; ++j) {
        const int nn = n0 + j * 16 + col;
        const float boj = bo[nn], nj = nullf[nn];
#pragma unroll
        for (int r = 0; r < 8; ++r) {
            long g = (long)mtile * 16 + r + 8 * half;
            float cv = cover[g];
            OUT[g * DD + nn] = c[j][r] + cv * boj + (1.0f - cv) * nj + X[g * DD + nn];
        }
    }
}

// ---------------------------------------------------------------------------
extern "C" void kernel_launch(void* const* d_in, const int* in_sizes, int n_in,
                              void* d_out, int out_size, void* d_ws, size_t ws_size,
                              hipStream_t stream) {
    const float* x     = (const float*)d_in[0];
    const float* masks = (const float*)d_in[1];
    const float* tf    = (const float*)d_in[2];
    const float* Wq    = (const float*)d_in[3];
    const float* Wk    = (const float*)d_in[4];
    const float* Wv    = (const float*)d_in[5];
    const float* Wo    = (const float*)d_in[6];
    const float* bo    = (const float*)d_in[7];
    const float* nullf = (const float*)d_in[8];
    float* out = (float*)d_out;

    char* ws = (char*)d_ws;
    size_t off = 0;
    auto alloc = [&](size_t bytes) -> void* {
        void* p = ws + off;
        off = (off + bytes + 255) & ~(size_t)255;
        return p;
    };
    _Float16* WqT = (_Float16*)alloc(2UL * DD * DD);
    _Float16* WkT = (_Float16*)alloc(2UL * DD * CTX);
    _Float16* WvT = (_Float16*)alloc(2UL * DD * CTX);
    _Float16* WoT = (_Float16*)alloc(2UL * DD * DD);
    _Float16* Qb  = (_Float16*)alloc(2UL * BB * LQ * DD);
    _Float16* Kb  = (_Float16*)alloc(2UL * BN * LTP * DD);
    _Float16* VTb = (_Float16*)alloc(2UL * BN * NH * DH * LTP);
    float*    ACC = (float*)alloc(4UL * BB * LQ * DD);
    float*    COV = (float*)alloc(4UL * BB * LQ);

    // 1) weight convert/transpose to f16 (B-operands become contiguous loads)
    {
        long t = (long)DD * DD;
        k_transpose_f16<<<(t + 255) / 256, 256, 0, stream>>>(Wq, WqT, DD, DD);
        k_transpose_f16<<<(t + 255) / 256, 256, 0, stream>>>(Wo, WoT, DD, DD);
        long t2 = (long)CTX * DD;
        k_transpose_f16<<<(t2 + 255) / 256, 256, 0, stream>>>(Wk, WkT, CTX, DD);
        k_transpose_f16<<<(t2 + 255) / 256, 256, 0, stream>>>(Wv, WvT, CTX, DD);
    }
    // 2) zero V^T (padded keys 77..95 must be exactly 0 for the AV product)
    {
        long t = (long)BN * NH * DH * LTP;
        k_zero_f16<<<(t + 255) / 256, 256, 0, stream>>>(VTb, t);
    }
    // 3) Q projection (once per batch: x is identical across objects)
    k_gemm_q<<<dim3(64, 16), 256, 0, stream>>>(x, WqT, Qb);
    // 4) K/V projections into attention layouts
    k_gemm_kv<<<dim3(10, 16), 256, 0, stream>>>(tf, WkT, Kb, VTb, 0);
    k_gemm_kv<<<dim3(10, 16), 256, 0, stream>>>(tf, WvT, Kb, VTb, 1);
    // 5) mask coverage
    k_cover<<<(BB * LQ + 255) / 256, 256, 0, stream>>>(masks, COV);
    // 6) fused attention + masked object-sum (2*16*256 waves, 4 waves/block)
    k_attn<<<2048, 128, 0, stream>>>(Qb, Kb, VTb, masks, ACC);
    // 7) output projection + epilogue (Wo pushed after the n-reduction)
    k_gemm_out<<<dim3(64, 16), 256, 0, stream>>>(ACC, WoT, x, bo, nullf, COV, out);
}
